// CRF_35304631173964
// MI455X (gfx1250) — compile-verified, matrix operand order checked by reading, not run on hardware
//
#include <hip/hip_runtime.h>
#include <hip/hip_bf16.h>

// CRF forward (partition) via per-step GEMM on exp-domain:
//   alpha' = m + log( exp(Tr) @ exp(alpha - m) ) + logit_t
// exp(Tr) is a constant 52x52 matrix -> padded to 64x64 bf16, consumed by
// v_wmma_f32_16x16x32_bf16 (4 M-tiles x 2 K-steps = 8 WMMAs / timestep / 16 batches).
// Branch-free inner loop; pad rows (52..63) ride as -inf through the recurrence
// (exp(Tr) pad rows are zero -> c=0 -> log->-inf -> exp->0), so no per-step mask.

#define KK 52
#define KP 64
#define TT 512
#define BB 1024
#define NEG_INF (-10000.0f)
#define START_TAG 50
#define STOP_TAG 51
#define LOG2E 1.4426950408889634f
#define LN2   0.6931471805599453f

typedef __attribute__((ext_vector_type(16))) __bf16        v16bf;
typedef __attribute__((ext_vector_type(8)))  float         v8f;
typedef __attribute__((ext_vector_type(8)))  unsigned int  v8u;

union Frag { v8u u; v16bf b; };

__device__ __forceinline__ unsigned int bf16_rne(float x) {
    unsigned int u = __float_as_uint(x);
    return (u + 0x7FFFu + ((u >> 16) & 1u)) >> 16;   // round-to-nearest-even
}
// Pack two f32 -> packed bf16 pair: round-half-up bias then one v_perm_b32
// grabbing the two high halves. 3 VALU ops total.
__device__ __forceinline__ unsigned int pk_bf16(float lo, float hi) {
    unsigned lu = __float_as_uint(lo) + 0x8000u;
    unsigned hu = __float_as_uint(hi) + 0x8000u;
    return __builtin_amdgcn_perm(hu, lu, 0x07060302u); // {hu[31:16], lu[31:16]}
}

// ---- prep: M16[i][j] = bf16(exp(Tr[i][j])), zero-padded to 64x64 -------------
__global__ void crf_prep_kernel(const float* __restrict__ trans,
                                unsigned short* __restrict__ M16) {
    for (int idx = threadIdx.x; idx < KP * KP; idx += blockDim.x) {
        int r = idx >> 6, c = idx & 63;
        float v = 0.0f;
        if (r < KK && c < KK) v = __expf(trans[r * KK + c]);
        M16[idx] = (unsigned short)bf16_rne(v);
    }
}

// ---- main: one wave (32 lanes) per 16-batch tile ----------------------------
__global__ __launch_bounds__(32)
void crf_fwd_kernel(const float* __restrict__ logits,
                    const float* __restrict__ trans,
                    const int*   __restrict__ labels,
                    const int*   __restrict__ lens,
                    const unsigned short* __restrict__ M16,
                    float* __restrict__ out)
{
    const int lane = threadIdx.x;
    const int col  = lane & 15;     // batch within tile == WMMA column
    const int half = lane >> 4;     // lane-half selects rows r vs r+8 / K vs K+8
    const int b    = blockIdx.x * 16 + col;
    const int len_b = lens[b];

    // Preload loop-invariant A fragments: exp(Tr) 64x64 bf16.
    // A layout (16-bit 16x32): lane holds row M = lane&15; low lanes K {0..7,16..23},
    // high lanes K {8..15,24..31} within each 32-wide K-step.
    Frag A[4][2];
#pragma unroll
    for (int mt = 0; mt < 4; ++mt) {
        const int arow = mt * 16 + col;
#pragma unroll
        for (int ks = 0; ks < 2; ++ks) {
#pragma unroll
            for (int p = 0; p < 8; ++p) {
                int kk = ks * 32 + ((p >> 2) << 4) + half * 8 + ((p & 3) << 1);
                unsigned lo = M16[arow * KP + kk];
                unsigned hi = M16[arow * KP + kk + 1];
                A[mt][ks].u[p] = lo | (hi << 16);
            }
        }
    }

    // alpha in C/D layout: alpha[mt][r] = row (mt*16 + half*8 + r), column = col.
    // Pad rows start at -inf and stay -inf without any per-step masking.
    float alpha[4][8];
#pragma unroll
    for (int mt = 0; mt < 4; ++mt)
#pragma unroll
        for (int r = 0; r < 8; ++r) {
            int row = mt * 16 + half * 8 + r;
            alpha[mt][r] = (row == START_TAG) ? 0.0f
                         : (row < KK)         ? NEG_INF
                                              : -__builtin_inff();
        }

    float em = 0.0f, ts = 0.0f;  // path scores; duplicated in both lane halves
    int prevLbl = START_TAG, lastLbl = 0;
    const size_t brow = (size_t)b * TT;

    for (int t = 0; t < TT; ++t) {
        const float* lp = logits + (brow + t) * KK;
        __builtin_prefetch(lp + 4 * KK, 0, 0);   // ~4 steps ahead -> global_prefetch_b8

        const bool act = (t < len_b);

        // ---- gather path scores, branch-free (both halves duplicate) ----
        {
            int   lbl = labels[brow + t];
            float e   = lp[lbl];                     // emission logit
            float tr  = trans[lbl * KK + prevLbl];   // Tr[nxt, prev]
            em      = act ? (em + e)  : em;
            ts      = act ? (ts + tr) : ts;
            lastLbl = act ? lbl : lastLbl;
            prevLbl = act ? lbl : prevLbl;
        }

        // ---- load this step's logits into D-fragment layout ----
        // base (b*T+t)*52*4 is 16B aligned (208 % 16 == 0); tags 48..51 cover tile 3,
        // pad rows (>=52) are -inf-dominated so q3 garbage reuse is fine.
        float lg[4][8];
#pragma unroll
        for (int mt = 0; mt < 3; ++mt) {
            const float4 q0 = *(const float4*)(lp + mt * 16 + half * 8 + 0);
            const float4 q1 = *(const float4*)(lp + mt * 16 + half * 8 + 4);
            lg[mt][0] = q0.x; lg[mt][1] = q0.y; lg[mt][2] = q0.z; lg[mt][3] = q0.w;
            lg[mt][4] = q1.x; lg[mt][5] = q1.y; lg[mt][6] = q1.z; lg[mt][7] = q1.w;
        }
        {
            const float4 q3 = *(const float4*)(lp + 48);
            const float qa[4] = {q3.x, q3.y, q3.z, q3.w};
#pragma unroll
            for (int r = 0; r < 8; ++r) lg[3][r] = qa[r & 3];
        }

        // ---- per-column max for numerically safe exp (pad rows are -inf) ----
        float mloc = alpha[0][0];
#pragma unroll
        for (int mt = 0; mt < 4; ++mt)
#pragma unroll
            for (int r = 0; r < 8; ++r) mloc = fmaxf(mloc, alpha[mt][r]);
        const float mcol = fmaxf(mloc, __shfl_xor(mloc, 16));

        // ---- exp (raw v_exp_f32, base 2) + lane-local repack into B frags ----
        // D layout (rows in VGPRs, col in lane, high lanes rows+8) is congruent
        // with the bf16 B layout (K pairs in VGPRs, col in lane, high lanes K+8).
        float ex[4][8];
#pragma unroll
        for (int mt = 0; mt < 4; ++mt)
#pragma unroll
            for (int r = 0; r < 8; ++r)
                ex[mt][r] = __builtin_amdgcn_exp2f((alpha[mt][r] - mcol) * LOG2E);

        Frag Bf[2];
#pragma unroll
        for (int p = 0; p < 4; ++p) {
            Bf[0].u[p]     = pk_bf16(ex[0][2 * p], ex[0][2 * p + 1]);
            Bf[0].u[p + 4] = pk_bf16(ex[1][2 * p], ex[1][2 * p + 1]);
            Bf[1].u[p]     = pk_bf16(ex[2][2 * p], ex[2][2 * p + 1]);
            Bf[1].u[p + 4] = pk_bf16(ex[3][2 * p], ex[3][2 * p + 1]);
        }

        // ---- GEMM: 4 M-tiles x 2 K-steps of v_wmma_f32_16x16x32_bf16 ----
#pragma unroll
        for (int mt = 0; mt < 4; ++mt) {
            v8f c = {0.f, 0.f, 0.f, 0.f, 0.f, 0.f, 0.f, 0.f};
            c = __builtin_amdgcn_wmma_f32_16x16x32_bf16(
                    false, A[mt][0].b, false, Bf[0].b, (short)0, c, false, false);
            c = __builtin_amdgcn_wmma_f32_16x16x32_bf16(
                    false, A[mt][1].b, false, Bf[1].b, (short)0, c, false, false);
#pragma unroll
            for (int r = 0; r < 8; ++r) {
                // raw v_log_f32 (log2): real rows give args in [~0.01, ~5e3];
                // pad rows give log(0) = -inf which is exactly what we want.
                float nv = mcol + __builtin_amdgcn_logf(c[r]) * LN2 + lg[mt][r];
                alpha[mt][r] = act ? nv : alpha[mt][r];  // step mask only
            }
        }
    }

    // ---- finalize ----
    ts += trans[STOP_TAG * KK + lastLbl];   // Tr[STOP, last]

#pragma unroll
    for (int mt = 0; mt < 4; ++mt)
#pragma unroll
        for (int r = 0; r < 8; ++r) {
            int row = mt * 16 + half * 8 + r;
            if (row < KK) alpha[mt][r] += trans[STOP_TAG * KK + row];  // guard = no OOB gather
        }

    float m2l = alpha[0][0];
#pragma unroll
    for (int mt = 0; mt < 4; ++mt)
#pragma unroll
        for (int r = 0; r < 8; ++r) m2l = fmaxf(m2l, alpha[mt][r]);
    const float m2 = fmaxf(m2l, __shfl_xor(m2l, 16));

    float s = 0.0f;
#pragma unroll
    for (int mt = 0; mt < 4; ++mt)
#pragma unroll
        for (int r = 0; r < 8; ++r)
            s += __builtin_amdgcn_exp2f((alpha[mt][r] - m2) * LOG2E);  // pad rows add 0
    s += __shfl_xor(s, 16);

    const float partition = m2 + __builtin_amdgcn_logf(s) * LN2;
    if (half == 0) out[b] = partition + em - ts;
}

extern "C" void kernel_launch(void* const* d_in, const int* in_sizes, int n_in,
                              void* d_out, int out_size, void* d_ws, size_t ws_size,
                              hipStream_t stream) {
    const float* logits = (const float*)d_in[0];   // (1024, 512, 52) f32
    const float* trans  = (const float*)d_in[1];   // (52, 52) f32
    const int*   labels = (const int*)d_in[2];     // (1024, 512) i32
    const int*   lens   = (const int*)d_in[3];     // (1024,) i32
    float* out = (float*)d_out;                    // (1024,) f32
    unsigned short* M16 = (unsigned short*)d_ws;   // 64x64 bf16 = 8 KB scratch

    crf_prep_kernel<<<1, 256, 0, stream>>>(trans, M16);
    crf_fwd_kernel<<<BB / 16, 32, 0, stream>>>(logits, trans, labels, lens, M16, out);
}
